// CrossAttentionBlock_2937757630974
// MI455X (gfx1250) — compile-verified, matrix operand order checked by reading, not run on hardware
//
#include <hip/hip_runtime.h>
#include <hip/hip_bf16.h>

// ---------------------------------------------------------------------------
// CrossAttentionBlock for MI455X (gfx1250, wave32, WMMA 16x16x32 f16)
// B=8, H=W=64 (N=4096 pixels), C=U=128.
// ---------------------------------------------------------------------------

typedef _Float16 f16;
typedef __attribute__((ext_vector_type(4)))  _Float16 v4h;
typedef __attribute__((ext_vector_type(8)))  _Float16 v8h;
typedef __attribute__((ext_vector_type(16))) _Float16 v16h;
typedef __attribute__((ext_vector_type(8)))  float    v8f;

#define NPIX   4096          // H*W
#define CDIM   128           // channels == units
#define NBATCH 8
#define MROWS  (NBATCH * NPIX)  // 32768 total pixel rows

// Load one 16x32 f16 A/B fragment for v_wmma_f32_16x16x32_f16.
// Per ISA layout: lane's 16 halves are two contiguous groups of 8 at
// k = half*8 .. +7 and k = 16+half*8 .. +7 (caller pre-offsets by half*8).
__device__ inline v16h load_frag16(const f16* p) {
  v8h lo = *(const v8h*)(p);
  v8h hi = *(const v8h*)(p + 16);
  return __builtin_shufflevector(lo, hi, 0, 1, 2, 3, 4, 5, 6, 7,
                                         8, 9, 10, 11, 12, 13, 14, 15);
}

__device__ inline v8f wmma16x16x32(v16h a, v16h b, v8f c) {
  // (neg_a, A, neg_b, B, c_mod, C, reuse_a, reuse_b)
  return __builtin_amdgcn_wmma_f32_16x16x32_f16(false, a, false, b,
                                                (short)0, c, false, false);
}

// Row reductions across the 16-lane half-wave group (xor masks 1,2,4,8 stay
// inside the group). Every lane in the group ends with the reduced value.
__device__ inline float rowmax16(float v) {
#pragma unroll
  for (int m = 8; m >= 1; m >>= 1) v = fmaxf(v, __shfl_xor(v, m, 32));
  return v;
}
__device__ inline float rowsum16(float v) {
#pragma unroll
  for (int m = 8; m >= 1; m >>= 1) v += __shfl_xor(v, m, 32);
  return v;
}

// ---------------------------------------------------------------------------
// Kernel 1: LayerNorm over C (eps=1e-3) + f16 conversion of xn and context.
// One wave per pixel row; 4 channels per lane; wave32 shuffle reduction.
// xn (f32) goes straight into d_out as the residual base.
// ---------------------------------------------------------------------------
__global__ __launch_bounds__(256) void ln_convert_kernel(
    const float* __restrict__ x, const float* __restrict__ ctx,
    const float* __restrict__ gamma, const float* __restrict__ beta,
    float* __restrict__ xn_out, f16* __restrict__ xnh, f16* __restrict__ ctxh) {
  const int lane = threadIdx.x & 31;
  const int wave = threadIdx.x >> 5;
  const size_t row = (size_t)blockIdx.x * 8 + wave;

  float4 xv = ((const float4*)(x + row * CDIM))[lane];
  float s = xv.x + xv.y + xv.z + xv.w;
#pragma unroll
  for (int m = 16; m >= 1; m >>= 1) s += __shfl_xor(s, m, 32);
  const float mu = s * (1.0f / 128.0f);

  float4 d;
  d.x = xv.x - mu; d.y = xv.y - mu; d.z = xv.z - mu; d.w = xv.w - mu;
  float vv = d.x * d.x + d.y * d.y + d.z * d.z + d.w * d.w;
#pragma unroll
  for (int m = 16; m >= 1; m >>= 1) vv += __shfl_xor(vv, m, 32);
  const float inv = rsqrtf(vv * (1.0f / 128.0f) + 1e-3f);

  float4 g  = ((const float4*)gamma)[lane];
  float4 bt = ((const float4*)beta)[lane];
  float4 o;
  o.x = d.x * inv * g.x + bt.x;
  o.y = d.y * inv * g.y + bt.y;
  o.z = d.z * inv * g.z + bt.z;
  o.w = d.w * inv * g.w + bt.w;

  ((float4*)(xn_out + row * CDIM))[lane] = o;
  v4h oh = {(f16)o.x, (f16)o.y, (f16)o.z, (f16)o.w};
  ((v4h*)(xnh + row * CDIM))[lane] = oh;

  float4 cv = ((const float4*)(ctx + row * CDIM))[lane];
  v4h ch = {(f16)cv.x, (f16)cv.y, (f16)cv.z, (f16)cv.w};
  ((v4h*)(ctxh + row * CDIM))[lane] = ch;
}

// ---------------------------------------------------------------------------
// Kernel 2: C[M x 128] = A[M x 128](f16) @ W[128 x 128](f32) + bias, via WMMA.
// 256 threads = 8 waves; tile = 128 rows x 128 cols; each wave owns 16 rows.
// W is staged in LDS transposed as f16 so B-fragments are two b128 ds loads.
// mode 0: f16 row-major [M,128]   (Q / K)
// mode 1: f16 transposed [b][c][n] (V, for fast PV B-fragments)
// mode 2: f32 accumulate into out_f (projection + residual)
// ---------------------------------------------------------------------------
__global__ __launch_bounds__(256) void gemm128_kernel(
    const f16* __restrict__ A, const float* __restrict__ W,
    const float* __restrict__ bias, f16* __restrict__ out_h,
    float* __restrict__ out_f, int mode) {
  __shared__ __attribute__((aligned(16))) f16 Wt[CDIM * CDIM];  // [n][k], 32 KB
  for (int idx = threadIdx.x; idx < CDIM * CDIM; idx += 256) {
    int k = idx >> 7, n = idx & 127;
    Wt[n * CDIM + k] = (f16)W[idx];
  }
  __syncthreads();

  const int lane = threadIdx.x & 31;
  const int wave = threadIdx.x >> 5;
  const int half = lane >> 4;
  const int l16  = lane & 15;
  const size_t row0 = (size_t)blockIdx.x * 128 + wave * 16;

  v16h a[4];
#pragma unroll
  for (int kk = 0; kk < 4; ++kk)
    a[kk] = load_frag16(A + (row0 + l16) * CDIM + kk * 32 + half * 8);

  v8f acc[8] = {};
#pragma unroll
  for (int kk = 0; kk < 4; ++kk) {
#pragma unroll
    for (int cn = 0; cn < 8; ++cn) {
      v16h bfrag = load_frag16(&Wt[(cn * 16 + l16) * CDIM + kk * 32 + half * 8]);
      acc[cn] = wmma16x16x32(a[kk], bfrag, acc[cn]);
    }
  }

#pragma unroll
  for (int cn = 0; cn < 8; ++cn) {
    const int n = cn * 16 + l16;
    const float bv = bias[n];
#pragma unroll
    for (int r = 0; r < 8; ++r) {
      const size_t grow = row0 + r + 8 * half;   // global row for C reg r
      const float val = acc[cn][r] + bv;
      if (mode == 0) {
        out_h[grow * CDIM + n] = (f16)val;
      } else if (mode == 1) {
        const size_t b = grow >> 12, nn = grow & (NPIX - 1);
        out_h[(b * CDIM + n) * NPIX + nn] = (f16)val;
      } else {
        out_f[grow * CDIM + n] += val;           // xn residual already there
      }
    }
  }
}

// ---------------------------------------------------------------------------
// Kernel 3: flash attention. 128 threads = 4 waves; 16 query rows per wave.
// KV streamed in blocks of 32: S = Q K^T (8 WMMAs), online softmax, P bounced
// through a wave-private LDS tile to re-layout C->A, then O += P V (8 WMMAs).
// ---------------------------------------------------------------------------
__global__ __launch_bounds__(128) void flash_attn_kernel(
    const f16* __restrict__ Q, const f16* __restrict__ K,
    const f16* __restrict__ Vt, f16* __restrict__ O) {
  __shared__ __attribute__((aligned(16))) f16 Pbuf[4][16 * 32];
  const int lane = threadIdx.x & 31;
  const int wave = threadIdx.x >> 5;
  const int half = lane >> 4;
  const int l16  = lane & 15;
  const int b    = blockIdx.x >> 6;
  const int q0   = (blockIdx.x & 63) * 64 + wave * 16;
  const size_t qbase = (size_t)b * NPIX + q0;
  const float scale = 0.08838834764831845f;  // U^-0.5, U=128

  v16h qa[4];
#pragma unroll
  for (int kk = 0; kk < 4; ++kk)
    qa[kk] = load_frag16(Q + (qbase + l16) * CDIM + kk * 32 + half * 8);

  v8f acc[8] = {};
  float mi[8], li[8];
#pragma unroll
  for (int r = 0; r < 8; ++r) { mi[r] = -3.0e38f; li[r] = 0.0f; }

  f16* Pl = &Pbuf[wave][0];
  const f16* Vbase = Vt + (size_t)b * CDIM * NPIX;

  for (int kv0 = 0; kv0 < NPIX; kv0 += 32) {
    // ---- S = Q K^T for 32 keys: two 16x16 f32 tiles --------------------
    v8f s0 = {}, s1 = {};
    const f16* Kb = K + ((size_t)b * NPIX + kv0) * CDIM;
#pragma unroll
    for (int kk = 0; kk < 4; ++kk) {
      v16h k0 = load_frag16(Kb + (size_t)l16 * CDIM + kk * 32 + half * 8);
      v16h k1 = load_frag16(Kb + (size_t)(l16 + 16) * CDIM + kk * 32 + half * 8);
      s0 = wmma16x16x32(qa[kk], k0, s0);
      s1 = wmma16x16x32(qa[kk], k1, s1);
    }

    // ---- online softmax (per C-layout row r) ---------------------------
#pragma unroll
    for (int r = 0; r < 8; ++r) {
      const float a0 = s0[r] * scale;
      const float a1 = s1[r] * scale;
      const float mblk = rowmax16(fmaxf(a0, a1));
      const float mnew = fmaxf(mi[r], mblk);
      const float corr = __expf(mi[r] - mnew);   // exp(-inf)=0 on first iter
      const float p0 = __expf(a0 - mnew);
      const float p1 = __expf(a1 - mnew);
      const float rs = rowsum16(p0 + p1);
      li[r] = li[r] * corr + rs;
      mi[r] = mnew;
#pragma unroll
      for (int cn = 0; cn < 8; ++cn) acc[cn][r] *= corr;
      const int prow = r + 8 * half;
      Pl[prow * 32 + l16]      = (f16)p0;
      Pl[prow * 32 + 16 + l16] = (f16)p1;
    }
    asm volatile("s_wait_dscnt 0" ::: "memory");  // wave-local LDS transpose

    // ---- O += P V : P as A-fragment (16x32), V^T gives contiguous B ----
    v16h pa = load_frag16(Pl + l16 * 32 + half * 8);
#pragma unroll
    for (int cn = 0; cn < 8; ++cn) {
      v16h vf = load_frag16(Vbase + (size_t)(cn * 16 + l16) * NPIX + kv0 + half * 8);
      acc[cn] = wmma16x16x32(pa, vf, acc[cn]);
    }
  }

  // ---- normalize and store f16 attention output ------------------------
#pragma unroll
  for (int r = 0; r < 8; ++r) {
    const float inv = 1.0f / li[r];
    const size_t grow = qbase + r + 8 * half;
#pragma unroll
    for (int cn = 0; cn < 8; ++cn)
      O[grow * CDIM + cn * 16 + l16] = (f16)(acc[cn][r] * inv);
  }
}

// ---------------------------------------------------------------------------
extern "C" void kernel_launch(void* const* d_in, const int* in_sizes, int n_in,
                              void* d_out, int out_size, void* d_ws, size_t ws_size,
                              hipStream_t stream) {
  (void)in_sizes; (void)n_in; (void)out_size; (void)ws_size;

  const float* inputs  = (const float*)d_in[0];
  const float* context = (const float*)d_in[1];
  const float* gamma   = (const float*)d_in[2];
  const float* beta    = (const float*)d_in[3];
  const float* Wq      = (const float*)d_in[4];
  const float* bq      = (const float*)d_in[5];
  const float* Wk      = (const float*)d_in[6];
  const float* bk      = (const float*)d_in[7];
  const float* Wv      = (const float*)d_in[8];
  const float* bv      = (const float*)d_in[9];
  const float* Wp      = (const float*)d_in[10];
  const float* bp      = (const float*)d_in[11];
  float* out = (float*)d_out;

  const size_t halfbuf = (size_t)MROWS * CDIM * sizeof(f16);  // 8 MB each
  char* ws = (char*)d_ws;
  f16* xnh  = (f16*)(ws + 0 * halfbuf);
  f16* ctxh = (f16*)(ws + 1 * halfbuf);
  f16* qh   = (f16*)(ws + 2 * halfbuf);
  f16* kh   = (f16*)(ws + 3 * halfbuf);
  f16* vt   = (f16*)(ws + 4 * halfbuf);   // [b][c][n] transposed
  f16* oh   = (f16*)(ws + 5 * halfbuf);

  // 1) LayerNorm + f16 conversions; xn (f32) -> d_out as residual base.
  ln_convert_kernel<<<MROWS / 8, 256, 0, stream>>>(inputs, context, gamma, beta,
                                                   out, xnh, ctxh);
  // 2) Q = xn @ Wq + bq ; K = ctx @ Wk + bk ; V^T = (ctx @ Wv + bv)^T
  gemm128_kernel<<<MROWS / 128, 256, 0, stream>>>(xnh,  Wq, bq, qh, nullptr, 0);
  gemm128_kernel<<<MROWS / 128, 256, 0, stream>>>(ctxh, Wk, bk, kh, nullptr, 0);
  gemm128_kernel<<<MROWS / 128, 256, 0, stream>>>(ctxh, Wv, bv, vt, nullptr, 1);
  // 3) fused softmax(Q K^T / sqrt(U)) V
  flash_attn_kernel<<<NBATCH * (NPIX / 64), 128, 0, stream>>>(qh, kh, vt, oh);
  // 4) d_out(=xn) += attn_out @ Wp + bp
  gemm128_kernel<<<MROWS / 128, 256, 0, stream>>>(oh, Wp, bp, nullptr, out, 2);
}
